// HierarchicalAttention_58574763983398
// MI455X (gfx1250) — compile-verified, matrix operand order checked by reading, not run on hardware
//
#include <hip/hip_runtime.h>
#include <hip/hip_bf16.h>
#include <math.h>

// ---------------------------------------------------------------------------
// CDNA5 (gfx1250, wave32) types and feature knobs
// ---------------------------------------------------------------------------
typedef __attribute__((ext_vector_type(16))) _Float16 v16h;
typedef __attribute__((ext_vector_type(8)))  _Float16 v8h;
typedef __attribute__((ext_vector_type(8)))  float    v8f;

#ifndef CDNA5_TRY_ASYNC
#define CDNA5_TRY_ASYNC 1   // use GLOBAL_LOAD_ASYNC_TO_LDS_B128 + s_wait_asynccnt
#endif

#define WMMA_F32_F16(A, B, C) \
  __builtin_amdgcn_wmma_f32_16x16x32_f16(false, (A), false, (B), (short)0, (C), false, false)

// Copy 16 f16 (32B) global -> LDS. Async variant is tracked by ASYNCcnt.
static __device__ __forceinline__ void stage16(_Float16* lds_dst, const _Float16* gsrc) {
#if CDNA5_TRY_ASYNC
  asm volatile("global_load_async_to_lds_b128 %0, %1, off"
               :: "v"((unsigned)(size_t)lds_dst), "v"(gsrc) : "memory");
  asm volatile("global_load_async_to_lds_b128 %0, %1, off"
               :: "v"((unsigned)(size_t)(lds_dst + 8)), "v"(gsrc + 8) : "memory");
#else
  *(v8h*)lds_dst = *(const v8h*)gsrc;
  *(v8h*)(lds_dst + 8) = *(const v8h*)(gsrc + 8);
#endif
}

static __device__ __forceinline__ void wait_async_le4() {
#if CDNA5_TRY_ASYNC
  asm volatile("s_wait_asynccnt 0x4" ::: "memory");
#endif
}
static __device__ __forceinline__ void wait_async_le0() {
#if CDNA5_TRY_ASYNC
  asm volatile("s_wait_asynccnt 0x0" ::: "memory");
#endif
}

// 16x32 16-bit WMMA A/B fragment from an LDS row (per ISA VGPR layout:
// lanes 0-15 hold K={0..7,16..23}, lanes 16-31 hold K={8..15,24..31}).
static __device__ __forceinline__ v16h lds_frag(const _Float16* rowptr, int lane) {
  const int kb = (lane & 16) ? 8 : 0;
  v8h lo = *(const v8h*)(rowptr + kb);
  v8h hi = *(const v8h*)(rowptr + kb + 16);
  return __builtin_shufflevector(lo, hi, 0, 1, 2, 3, 4, 5, 6, 7,
                                 8, 9, 10, 11, 12, 13, 14, 15);
}

static __device__ __forceinline__ float gelu_f(float x) {
  return 0.5f * x * (1.0f + erff(x * 0.70710678118654752f));
}

// ---------------------------------------------------------------------------
// GEMM: C[M,N] = A[M,K](f16) @ W[N,K](f16)^T + bias[N](f32), opt fused GELU.
// 128 threads / 4 waves; 64x64 block tile; K-step 32 = one WMMA K.
// Double-buffered LDS staging via async global->LDS copies.
// Dual epilogue: optional f32 and/or f16 outputs.
// ---------------------------------------------------------------------------
__global__ __launch_bounds__(128)
void gemm_kernel(const _Float16* __restrict__ A, const _Float16* __restrict__ W,
                 const float* __restrict__ bias, float* __restrict__ Cf,
                 _Float16* __restrict__ Ch, int M, int N, int K, int act) {
  __shared__ __align__(16) _Float16 As[2][64][40];   // 80B row stride
  __shared__ __align__(16) _Float16 Bs[2][64][40];

  const int tid  = threadIdx.x;
  const int lane = tid & 31;
  const int wave = tid >> 5;
  const int rowBase = blockIdx.y * 64;
  const int colBase = blockIdx.x * 64;
  const int mw = (wave >> 1) * 32;
  const int nw = (wave & 1) * 32;
  const int nl = lane & 15;
  const int mo = (lane >> 4) * 8;
  const int ldr = tid >> 1;          // staged row 0..63
  const int ldc = (tid & 1) * 16;    // staged col group

  int ar = rowBase + ldr; if (ar >= M) ar = M - 1;   // clamp ragged M
  const _Float16* ap = A + (size_t)ar * K + ldc;
  const _Float16* wp = W + (size_t)(colBase + ldr) * K + ldc;

  v8f acc[2][2];
#pragma unroll
  for (int i = 0; i < 2; ++i)
#pragma unroll
    for (int j = 0; j < 2; ++j)
#pragma unroll
      for (int e = 0; e < 8; ++e) acc[i][j][e] = 0.0f;

  const int T = K >> 5;
  // prologue: stage tile 0 into buffer 0
  stage16(&As[0][ldr][ldc], ap);
  stage16(&Bs[0][ldr][ldc], wp);

  for (int i = 0; i < T; ++i) {
    const int cur = i & 1;
    if (i + 1 < T) {   // prefetch next tile into alternate buffer
      stage16(&As[cur ^ 1][ldr][ldc], ap + (size_t)(i + 1) * 32);
      stage16(&Bs[cur ^ 1][ldr][ldc], wp + (size_t)(i + 1) * 32);
      wait_async_le4();   // tile i has landed (in-order completion)
    } else {
      wait_async_le0();
    }
    __syncthreads();

    v16h af[2], bf[2];
#pragma unroll
    for (int ii = 0; ii < 2; ++ii) af[ii] = lds_frag(&As[cur][mw + ii * 16 + nl][0], lane);
#pragma unroll
    for (int jj = 0; jj < 2; ++jj) bf[jj] = lds_frag(&Bs[cur][nw + jj * 16 + nl][0], lane);
#pragma unroll
    for (int ii = 0; ii < 2; ++ii)
#pragma unroll
      for (int jj = 0; jj < 2; ++jj)
        acc[ii][jj] = WMMA_F32_F16(af[ii], bf[jj], acc[ii][jj]);
    __syncthreads();
  }

  // C/D layout: lane = col (n & 15, halves), VGPR v = row (mo + v).
#pragma unroll
  for (int i = 0; i < 2; ++i)
#pragma unroll
    for (int j = 0; j < 2; ++j)
#pragma unroll
      for (int v = 0; v < 8; ++v) {
        int m = rowBase + mw + i * 16 + mo + v;
        int n = colBase + nw + j * 16 + nl;
        if (m < M) {
          float r = acc[i][j][v] + bias[n];
          if (act) r = gelu_f(r);
          if (Cf) Cf[(size_t)m * N + n] = r;
          if (Ch) Ch[(size_t)m * N + n] = (_Float16)r;
        }
      }
}

// ---------------------------------------------------------------------------
// Flash attention. Q,K,V f16 [B,S,D], head h in cols [h*64, h*64+64).
// One (64-query tile, head, batch) block, 4 waves. Online softmax.
// K-tile staged by async copies (overlapped with manual V transpose).
// Output written as f16 (feeds the o-projection GEMM).
// ---------------------------------------------------------------------------
__global__ __launch_bounds__(128)
void flash_kernel(const _Float16* __restrict__ Q, const _Float16* __restrict__ Kg,
                  const _Float16* __restrict__ Vg, _Float16* __restrict__ Oh,
                  int Sq, int Skv, int Dm, float scale) {
  __shared__ __align__(16) _Float16 Ks[32][72];      // 32 keys x 64 dk (pad)
  __shared__ __align__(16) _Float16 Vt[64][40];      // transposed: dk x keys
  __shared__ __align__(16) _Float16 Ps[4][16][40];   // per-wave P scratch

  const int tid  = threadIdx.x;
  const int lane = tid & 31;
  const int wave = tid >> 5;
  const int h = blockIdx.y;
  const int b = blockIdx.z;
  const int qbase = blockIdx.x * 64 + wave * 16;
  const int nl = lane & 15;
  const int mo = (lane >> 4) * 8;
  const int kb = (lane & 16) ? 8 : 0;

  // Q fragments: two dk-halves, read straight from global f16 (16B aligned)
  int qrow = qbase + nl; if (qrow >= Sq) qrow = Sq - 1;
  const _Float16* qp = Q + ((size_t)(b * Sq + qrow)) * Dm + h * 64;
  v16h qf[2];
#pragma unroll
  for (int hh = 0; hh < 2; ++hh) {
    const _Float16* p = qp + 32 * hh + kb;
    v8h lo = *(const v8h*)p;
    v8h hi = *(const v8h*)(p + 16);
    qf[hh] = __builtin_shufflevector(lo, hi, 0, 1, 2, 3, 4, 5, 6, 7,
                                     8, 9, 10, 11, 12, 13, 14, 15);
  }

  v8f oacc[4];
#pragma unroll
  for (int d = 0; d < 4; ++d)
#pragma unroll
    for (int e = 0; e < 8; ++e) oacc[d][e] = 0.0f;
  float mrun[8], lrun[8];
#pragma unroll
  for (int v = 0; v < 8; ++v) { mrun[v] = -1.0e30f; lrun[v] = 0.0f; }

  const int lr = tid >> 2;           // key row staged by this thread
  const int lc = (tid & 3) * 16;     // col group

  for (int kv = 0; kv < Skv; kv += 32) {
    const _Float16* kp = Kg + ((size_t)(b * Skv + kv + lr)) * Dm + h * 64 + lc;
    stage16(&Ks[lr][lc], kp);                         // async K-tile copy
    const _Float16* vp = Vg + ((size_t)(b * Skv + kv + lr)) * Dm + h * 64 + lc;
    v8h v0 = *(const v8h*)vp;                         // V transpose (overlaps)
    v8h v1 = *(const v8h*)(vp + 8);
#pragma unroll
    for (int i = 0; i < 8; ++i) { Vt[lc + i][lr] = v0[i]; Vt[lc + 8 + i][lr] = v1[i]; }
    wait_async_le0();
    __syncthreads();

    // scores S = Q(16x64) @ K^T(64x32): 2 key n-tiles x 2 dk-halves
    v8f s0, s1;
#pragma unroll
    for (int e = 0; e < 8; ++e) { s0[e] = 0.0f; s1[e] = 0.0f; }
#pragma unroll
    for (int hh = 0; hh < 2; ++hh) {
      v16h k0f = lds_frag(&Ks[nl][32 * hh], lane);
      v16h k1f = lds_frag(&Ks[16 + nl][32 * hh], lane);
      s0 = WMMA_F32_F16(qf[hh], k0f, s0);
      s1 = WMMA_F32_F16(qf[hh], k1f, s1);
    }

    // online softmax: each row lives across one 16-lane half
#pragma unroll
    for (int v = 0; v < 8; ++v) {
      float a0 = s0[v] * scale;
      float a1 = s1[v] * scale;
      float mx = fmaxf(a0, a1);
#pragma unroll
      for (int msk = 1; msk <= 8; msk <<= 1) mx = fmaxf(mx, __shfl_xor(mx, msk, 32));
      float mnew  = fmaxf(mrun[v], mx);
      float alpha = __expf(mrun[v] - mnew);
      mrun[v] = mnew;
      float p0 = __expf(a0 - mnew);
      float p1 = __expf(a1 - mnew);
      float rs = p0 + p1;
#pragma unroll
      for (int msk = 1; msk <= 8; msk <<= 1) rs += __shfl_xor(rs, msk, 32);
      lrun[v] = lrun[v] * alpha + rs;
#pragma unroll
      for (int d = 0; d < 4; ++d) oacc[d][v] *= alpha;
      Ps[wave][mo + v][nl]      = (_Float16)p0;
      Ps[wave][mo + v][16 + nl] = (_Float16)p1;
    }

    // O += P(16x32) @ V(32x64): 4 dk n-tiles
    v16h pf = lds_frag(&Ps[wave][nl][0], lane);
#pragma unroll
    for (int d = 0; d < 4; ++d) {
      v16h vf = lds_frag(&Vt[d * 16 + nl][0], lane);
      oacc[d] = WMMA_F32_F16(pf, vf, oacc[d]);
    }
    __syncthreads();
  }

#pragma unroll
  for (int v = 0; v < 8; ++v) {
    int m = qbase + mo + v;
    if (m < Sq) {
      float inv = 1.0f / lrun[v];
#pragma unroll
      for (int d = 0; d < 4; ++d)
        Oh[((size_t)(b * Sq + m)) * Dm + h * 64 + d * 16 + nl] = (_Float16)(oacc[d][v] * inv);
    }
  }
}

// ---------------------------------------------------------------------------
// Elementwise / reduction helpers (f32 math, optional f16 side outputs)
// ---------------------------------------------------------------------------
__global__ void cast_kernel(const float* __restrict__ src, _Float16* __restrict__ dst, long n) {
  long i = (long)blockIdx.x * blockDim.x + threadIdx.x;
  long stride = (long)gridDim.x * blockDim.x;
  for (; i < n; i += stride) dst[i] = (_Float16)src[i];
}

__global__ __launch_bounds__(256)
void add_ln_kernel(const float* __restrict__ a, const float* __restrict__ c,
                   const float* __restrict__ g, const float* __restrict__ beta,
                   float* __restrict__ out, _Float16* __restrict__ outh, int D) {
  __shared__ float red[256];
  const size_t base = (size_t)blockIdx.x * D;
  float s = 0.0f;
  for (int d = threadIdx.x; d < D; d += 256) s += a[base + d] + c[base + d];
  red[threadIdx.x] = s; __syncthreads();
  for (int off = 128; off > 0; off >>= 1) {
    if (threadIdx.x < off) red[threadIdx.x] += red[threadIdx.x + off];
    __syncthreads();
  }
  const float mu = red[0] / (float)D;
  __syncthreads();
  float q = 0.0f;
  for (int d = threadIdx.x; d < D; d += 256) {
    float t = a[base + d] + c[base + d] - mu; q += t * t;
  }
  red[threadIdx.x] = q; __syncthreads();
  for (int off = 128; off > 0; off >>= 1) {
    if (threadIdx.x < off) red[threadIdx.x] += red[threadIdx.x + off];
    __syncthreads();
  }
  const float rstd = rsqrtf(red[0] / (float)D + 1e-5f);
  for (int d = threadIdx.x; d < D; d += 256) {
    float r = (a[base + d] + c[base + d] - mu) * rstd * g[d] + beta[d];
    out[base + d] = r;
    if (outh) outh[base + d] = (_Float16)r;
  }
}

__global__ void add_kernel(const float* a, const float* b, float* out,
                           _Float16* outh, long n) {
  long i = (long)blockIdx.x * blockDim.x + threadIdx.x;
  long stride = (long)gridDim.x * blockDim.x;
  for (; i < n; i += stride) {
    float r = a[i] + b[i];
    out[i] = r;
    if (outh) outh[i] = (_Float16)r;
  }
}

__global__ void seg_mean_kernel(const float* __restrict__ x, const int* __restrict__ cb,
                                float* __restrict__ cr, _Float16* __restrict__ crh,
                                int S, int D, int C) {
  const int bc = blockIdx.x;                 // b*C + c
  const int b  = bc / C;
  const int s0 = cb[bc * 2], s1 = cb[bc * 2 + 1];
  const float inv = 1.0f / (float)((s1 - s0) > 0 ? (s1 - s0) : 1);
  for (int d = threadIdx.x; d < D; d += blockDim.x) {
    float acc = 0.0f;
    for (int s = s0; s < s1; ++s) acc += x[((size_t)(b * S + s)) * D + d];
    float r = acc * inv;
    cr[(size_t)bc * D + d] = r;
    crh[(size_t)bc * D + d] = (_Float16)r;
  }
}

__global__ void scatter_add_kernel(float* __restrict__ x, const float* __restrict__ cr,
                                   const int* __restrict__ cb, int S, int D, int C) {
  const int bc = blockIdx.x;
  const int b  = bc / C;
  const int s0 = cb[bc * 2], s1 = cb[bc * 2 + 1];
  const int span = (s1 - s0) * D;
  for (int idx = threadIdx.x; idx < span; idx += blockDim.x) {
    int s = s0 + idx / D, d = idx % D;
    x[((size_t)(b * S + s)) * D + d] += cr[(size_t)bc * D + d];
  }
}

__global__ void doc_mean_kernel(const float* __restrict__ x, float* __restrict__ doc,
                                _Float16* __restrict__ doch, int S, int D) {
  const int b = blockIdx.x;
  const float inv = 1.0f / (float)S;
  for (int d = threadIdx.x; d < D; d += blockDim.x) {
    float acc = 0.0f;
    for (int s = 0; s < S; ++s) acc += x[((size_t)(b * S + s)) * D + d];
    float r = acc * inv;
    doc[(size_t)b * D + d] = r;
    doch[(size_t)b * D + d] = (_Float16)r;
  }
}

__global__ void broadcast_add_kernel(const float* __restrict__ x, const float* __restrict__ doc,
                                     float* __restrict__ out, int S, int D, long n) {
  long i = (long)blockIdx.x * blockDim.x + threadIdx.x;
  long stride = (long)gridDim.x * blockDim.x;
  const long SD = (long)S * D;
  for (; i < n; i += stride) {
    long b = i / SD;
    long d = i % D;
    out[i] = x[i] + doc[b * D + d];
  }
}

// ---------------------------------------------------------------------------
// Host orchestration
// ---------------------------------------------------------------------------
static inline void launch_gemm(hipStream_t s, const _Float16* A, const _Float16* W,
                               const float* bias, float* Cf, _Float16* Ch,
                               int M, int N, int K, int act) {
  dim3 grid(N / 64, (M + 63) / 64), blk(128);
  gemm_kernel<<<grid, blk, 0, s>>>(A, W, bias, Cf, Ch, M, N, K, act);
}

extern "C" void kernel_launch(void* const* d_in, const int* in_sizes, int n_in,
                              void* d_out, int out_size, void* d_ws, size_t ws_size,
                              hipStream_t stream) {
  (void)in_sizes; (void)n_in; (void)out_size; (void)ws_size;
  const int Bb = 2, S = 2048, Dm = 1024, Hh = 16, Cc = 32, Df = 4096;
  const int BS = Bb * S;
  const float scale = 0.125f;  // 1/sqrt(64)

  const float* x  = (const float*)d_in[0];
  const int*   cb = (const int*)d_in[1];
  // params in setup_inputs() dict order: levels {tok, chk, doc}, each:
  // qw,qb,kw,kb,vw,vb,ow,ob, ln_g,ln_b, w1,b1,w2,b2  (14 tensors)
  auto P = [&](int level, int t) { return (const float*)d_in[2 + level * 14 + t]; };

  const size_t NA = (size_t)BS * Dm;       // 4,194,304
  const size_t CD = (size_t)Bb * Cc * Dm;  // 65,536
  const size_t CH = (size_t)Bb * Cc * Df;  // 262,144
  const size_t DD = (size_t)Bb * Dm;       // 2,048
  const size_t DH = (size_t)Bb * Df;       // 8,192
  const size_t DW = (size_t)Dm * Dm;       // attn weight
  const size_t FW = (size_t)Df * Dm;       // ffn weight

  char* wsp = (char*)d_ws;
  auto af32 = [&](size_t n) { float* p = (float*)wsp; wsp += n * sizeof(float); return p; };
  auto af16 = [&](size_t n) { _Float16* p = (_Float16*)wsp; wsp += n * sizeof(_Float16); return p; };

  // f32 buffers
  float* F0 = af32(NA);   // attn out / scratch
  float* F1 = af32(NA);   // xn
  float* F2 = af32(NA);   // ffn out
  float* F3 = af32(NA);   // xt (running token state)
  float* crf = af32(CD);  float* cof = af32(CD);  float* cnf = af32(CD);
  float* cff = af32(CD);  float* cr2 = af32(CD);
  float* dmf = af32(DD);  float* dOf = af32(DD);  float* dnf = af32(DD);
  float* dff = af32(DD);  float* dfin = af32(DD);

  // f16 buffers
  _Float16* xh  = af16(NA);
  _Float16* H0  = af16(NA);       // q  -> xn(h)
  _Float16* H1  = af16(NA);       // k  -> xt(h)
  _Float16* H2  = af16(NA);       // v  -> doc-k
  _Float16* H3  = af16(NA);       // ctx-> doc-v
  _Float16* H5  = af16(4 * NA);   // ffn hidden
  _Float16* crh = af16(CD);  _Float16* cqh = af16(CD);  _Float16* ckh = af16(CD);
  _Float16* cvh = af16(CD);  _Float16* cxh = af16(CD);  _Float16* cnh = af16(CD);
  _Float16* chh = af16(CH);
  _Float16* dmh = af16(DD);  _Float16* dqh = af16(DD);  _Float16* dxh = af16(DD);
  _Float16* dnh = af16(DD);  _Float16* dhh = af16(DH);
  // f16 weights: [level][qw,kw,vw,ow,w1,w2]
  _Float16* wh[3][6];
  for (int L = 0; L < 3; ++L) {
    wh[L][0] = af16(DW); wh[L][1] = af16(DW); wh[L][2] = af16(DW); wh[L][3] = af16(DW);
    wh[L][4] = af16(FW); wh[L][5] = af16(FW);
  }

  const int eb = 256, eg = 8192;

  // ---- pre-pass: cast activations + weights to f16 ----
  cast_kernel<<<eg, eb, 0, stream>>>(x, xh, (long)NA);
  const int widx[6] = {0, 2, 4, 6, 10, 12};
  for (int L = 0; L < 3; ++L)
    for (int t = 0; t < 6; ++t) {
      size_t n = (t < 4) ? DW : FW;
      cast_kernel<<<4096, eb, 0, stream>>>(P(L, widx[t]), wh[L][t], (long)n);
    }

  // ===================== token level =====================
  launch_gemm(stream, xh, wh[0][0], P(0, 1), nullptr, H0, BS, Dm, Dm, 0);  // Q
  launch_gemm(stream, xh, wh[0][1], P(0, 3), nullptr, H1, BS, Dm, Dm, 0);  // K
  launch_gemm(stream, xh, wh[0][2], P(0, 5), nullptr, H2, BS, Dm, Dm, 0);  // V
  flash_kernel<<<dim3(S / 64, Hh, Bb), 128, 0, stream>>>(H0, H1, H2, H3, S, S, Dm, scale);
  launch_gemm(stream, H3, wh[0][3], P(0, 7), F0, nullptr, BS, Dm, Dm, 0);  // attn out
  add_ln_kernel<<<BS, eb, 0, stream>>>(x, F0, P(0, 8), P(0, 9), F1, H0, Dm);   // xn
  launch_gemm(stream, H0, wh[0][4], P(0, 11), nullptr, H5, BS, Df, Dm, 1);     // h=gelu
  launch_gemm(stream, H5, wh[0][5], P(0, 13), F2, nullptr, BS, Dm, Df, 0);     // f
  add_kernel<<<eg, eb, 0, stream>>>(F1, F2, F3, H1, (long)NA);                 // xt

  // ===================== chunk level =====================
  seg_mean_kernel<<<Bb * Cc, eb, 0, stream>>>(F3, cb, crf, crh, S, Dm, Cc);
  launch_gemm(stream, crh, wh[1][0], P(1, 1), nullptr, cqh, Bb * Cc, Dm, Dm, 0);
  launch_gemm(stream, crh, wh[1][1], P(1, 3), nullptr, ckh, Bb * Cc, Dm, Dm, 0);
  launch_gemm(stream, crh, wh[1][2], P(1, 5), nullptr, cvh, Bb * Cc, Dm, Dm, 0);
  flash_kernel<<<dim3(1, Hh, Bb), 128, 0, stream>>>(cqh, ckh, cvh, cxh, Cc, Cc, Dm, scale);
  launch_gemm(stream, cxh, wh[1][3], P(1, 7), cof, nullptr, Bb * Cc, Dm, Dm, 0);
  add_ln_kernel<<<Bb * Cc, eb, 0, stream>>>(crf, cof, P(1, 8), P(1, 9), cnf, cnh, Dm);
  launch_gemm(stream, cnh, wh[1][4], P(1, 11), nullptr, chh, Bb * Cc, Df, Dm, 1);
  launch_gemm(stream, chh, wh[1][5], P(1, 13), cff, nullptr, Bb * Cc, Dm, Df, 0);
  add_kernel<<<256, eb, 0, stream>>>(cnf, cff, cr2, nullptr, (long)CD);
  scatter_add_kernel<<<Bb * Cc, eb, 0, stream>>>(F3, cr2, cb, S, Dm, Cc);
  cast_kernel<<<eg, eb, 0, stream>>>(F3, H1, (long)NA);   // refresh xt(f16)

  // ===================== document level =====================
  doc_mean_kernel<<<Bb, eb, 0, stream>>>(F3, dmf, dmh, S, Dm);
  launch_gemm(stream, dmh, wh[2][0], P(2, 1), nullptr, dqh, Bb, Dm, Dm, 0);
  launch_gemm(stream, H1, wh[2][1], P(2, 3), nullptr, H2, BS, Dm, Dm, 0);  // K(tokens)
  launch_gemm(stream, H1, wh[2][2], P(2, 5), nullptr, H3, BS, Dm, Dm, 0);  // V(tokens)
  flash_kernel<<<dim3(1, Hh, Bb), 128, 0, stream>>>(dqh, H2, H3, dxh, 1, S, Dm, scale);
  launch_gemm(stream, dxh, wh[2][3], P(2, 7), dOf, nullptr, Bb, Dm, Dm, 0);
  add_ln_kernel<<<Bb, eb, 0, stream>>>(dmf, dOf, P(2, 8), P(2, 9), dnf, dnh, Dm);
  launch_gemm(stream, dnh, wh[2][4], P(2, 11), nullptr, dhh, Bb, Df, Dm, 1);
  launch_gemm(stream, dhh, wh[2][5], P(2, 13), dff, nullptr, Bb, Dm, Df, 0);
  add_kernel<<<32, eb, 0, stream>>>(dnf, dff, dfin, nullptr, (long)DD);
  broadcast_add_kernel<<<eg, eb, 0, stream>>>(F3, dfin, (float*)d_out, S, Dm, (long)NA);
}
